// DIPNet_1236950581970
// MI455X (gfx1250) — matrix operand order repaired and checked by reading, not compiled, not run on hardware
//
#include <hip/hip_runtime.h>
#include <math.h>

typedef __attribute__((ext_vector_type(16))) _Float16 v16h;
typedef __attribute__((ext_vector_type(8)))  _Float16 v8h;
typedef __attribute__((ext_vector_type(8)))  float    v8f;

#define BS_      4
#define NPATCH_  256
#define DIM_     384
#define HID_     2688
#define NC_      1000
#define NCP_     1024      // class dim padded to 64-tile multiple
#define NDIST_   7
#define NKEY_    2048      // 8 * 256 distractor keys
#define GRID_    16
#define PW_      14
#define IMG_     224
#define NROW_    2048      // 2 * BS * NPATCH projector rows
#define INV_BETA 14.2857142857142857f

// ---------------- WMMA fragment loaders (ISA 7.12.2 layouts) ----------------
// All pointers below are 16-byte aligned by construction (row strides are
// multiples of 16 f16, k offsets are multiples of 8 f16).

// A (16x32 f16, M x K): lanes 0-15 -> row M=lane&15, elems 0..7 = K k0..k0+7,
// elems 8..15 = K k0+16..k0+23 ; lanes 16-31 -> K offset +8.
__device__ __forceinline__ v16h frag_a(const _Float16* __restrict__ Arow,
                                       int k0, int lane) {
  const _Float16* p = Arow + k0 + ((lane >> 4) << 3);
  v8h lo = *(const v8h*)p;
  v8h hi = *(const v8h*)(p + 16);
  return __builtin_shufflevector(lo, hi, 0, 1, 2, 3, 4, 5, 6, 7,
                                 8, 9, 10, 11, 12, 13, 14, 15);
}

// B (32x16 f16, K x N) where column N is a contiguous row of a transposed
// matrix: lanes 0-15 -> col, elems 0..7 = K k0..k0+7, elems 8..15 = K k0+16..k0+23 ;
// lanes 16-31 -> K offset +8.
__device__ __forceinline__ v16h frag_b(const _Float16* __restrict__ Bcol,
                                       int k0, int lane) {
  const _Float16* p = Bcol + k0 + ((lane >> 4) << 3);
  v8h lo = *(const v8h*)p;
  v8h hi = *(const v8h*)(p + 16);
  return __builtin_shufflevector(lo, hi, 0, 1, 2, 3, 4, 5, 6, 7,
                                 8, 9, 10, 11, 12, 13, 14, 15);
}

#define WMMA(acc, a, b) \
  acc = __builtin_amdgcn_wmma_f32_16x16x32_f16(false, a, false, b, (short)0, acc, \
                                               false, false)

// ---------------- converts ----------------

__global__ void k_cvt(const float* __restrict__ in, _Float16* __restrict__ out, int n) {
  int i = blockIdx.x * blockDim.x + threadIdx.x;
  if (i < n) out[i] = (_Float16)in[i];
}

// f32 (K x N) -> f16 transposed (N x K): B columns become contiguous rows.
__global__ void k_cvt_t(const float* __restrict__ in, _Float16* __restrict__ out,
                        int K, int N) {
  int i = blockIdx.x * blockDim.x + threadIdx.x;
  if (i >= K * N) return;
  int k = i / N, n = i - k * N;
  out[(size_t)n * K + k] = (_Float16)in[i];
}

// ---------------- GEMM1: H = GELU(X @ W1 + b1) ----------------
// X (2048x384), W1T (2688x384). Wave computes M=16, N=64 (4 accumulators).
// grid (21, 128), block 64 (2 waves).
__global__ void k_fc1_gelu(const _Float16* __restrict__ X, const _Float16* __restrict__ W1T,
                           const float* __restrict__ b1, _Float16* __restrict__ H) {
  const int lane = threadIdx.x & 31;
  const int wave = threadIdx.x >> 5;
  const int n0 = (blockIdx.x * 2 + wave) * 64;
  const int m0 = blockIdx.y * 16;
  const _Float16* Arow = X + (size_t)(m0 + (lane & 15)) * DIM_;
  const _Float16* B0 = W1T + (size_t)(n0 + (lane & 15)) * DIM_;
  const _Float16* B1 = B0 + (size_t)16 * DIM_;
  const _Float16* B2 = B0 + (size_t)32 * DIM_;
  const _Float16* B3 = B0 + (size_t)48 * DIM_;
  v8f acc0 = {0.f,0.f,0.f,0.f,0.f,0.f,0.f,0.f};
  v8f acc1 = acc0, acc2 = acc0, acc3 = acc0;
#pragma unroll
  for (int k0 = 0; k0 < DIM_; k0 += 32) {
    v16h a = frag_a(Arow, k0, lane);
    WMMA(acc0, a, frag_b(B0, k0, lane));
    WMMA(acc1, a, frag_b(B1, k0, lane));
    WMMA(acc2, a, frag_b(B2, k0, lane));
    WMMA(acc3, a, frag_b(B3, k0, lane));
  }
  const int mB = m0 + ((lane >> 4) << 3);
  const v8f* accs[4] = {&acc0, &acc1, &acc2, &acc3};
#pragma unroll
  for (int t = 0; t < 4; ++t) {
    const int col = n0 + t * 16 + (lane & 15);
    const float bias = b1[col];
#pragma unroll
    for (int i = 0; i < 8; ++i) {
      float x = (*accs[t])[i] + bias;
      float g = 0.5f * x * (1.0f + erff(x * 0.70710678118654752f));  // exact erf GELU
      H[(size_t)(mB + i) * HID_ + col] = (_Float16)g;
    }
  }
}

// ---------------- GEMM2: Y = H @ W2 (f32 out) ----------------
// H (2048x2688), W2T (384x2688). grid (6, 128), block 32.
__global__ void k_fc2(const _Float16* __restrict__ H, const _Float16* __restrict__ W2T,
                      float* __restrict__ Y) {
  const int lane = threadIdx.x & 31;
  const int n0 = blockIdx.x * 64;
  const int m0 = blockIdx.y * 16;
  const _Float16* Arow = H + (size_t)(m0 + (lane & 15)) * HID_;
  const _Float16* B0 = W2T + (size_t)(n0 + (lane & 15)) * HID_;
  const _Float16* B1 = B0 + (size_t)16 * HID_;
  const _Float16* B2 = B0 + (size_t)32 * HID_;
  const _Float16* B3 = B0 + (size_t)48 * HID_;
  v8f acc0 = {0.f,0.f,0.f,0.f,0.f,0.f,0.f,0.f};
  v8f acc1 = acc0, acc2 = acc0, acc3 = acc0;
#pragma unroll 2
  for (int k0 = 0; k0 < HID_; k0 += 32) {
    v16h a = frag_a(Arow, k0, lane);
    WMMA(acc0, a, frag_b(B0, k0, lane));
    WMMA(acc1, a, frag_b(B1, k0, lane));
    WMMA(acc2, a, frag_b(B2, k0, lane));
    WMMA(acc3, a, frag_b(B3, k0, lane));
  }
  const int mB = m0 + ((lane >> 4) << 3);
  const v8f* accs[4] = {&acc0, &acc1, &acc2, &acc3};
#pragma unroll
  for (int t = 0; t < 4; ++t) {
    const int col = n0 + t * 16 + (lane & 15);
#pragma unroll
    for (int i = 0; i < 8; ++i)
      Y[(size_t)(mB + i) * DIM_ + col] = (*accs[t])[i];
  }
}

// ---------------- LayerNorm + l2norm -> f16 q/k rows ----------------
__global__ void k_ln_l2(const float* __restrict__ Y, const float* __restrict__ b2,
                        const float* __restrict__ lnw, const float* __restrict__ lnb,
                        _Float16* __restrict__ QK) {
  __shared__ float red[128];
  const int t = threadIdx.x;
  const size_t row = blockIdx.x;
  const float* y = Y + row * DIM_;
  float x0 = y[t] + b2[t];
  float x1 = y[t + 128] + b2[t + 128];
  float x2 = y[t + 256] + b2[t + 256];

  red[t] = x0 + x1 + x2;
  __syncthreads();
  for (int o = 64; o > 0; o >>= 1) { if (t < o) red[t] += red[t + o]; __syncthreads(); }
  const float mu = red[0] * (1.0f / DIM_);
  __syncthreads();

  const float d0 = x0 - mu, d1 = x1 - mu, d2 = x2 - mu;
  red[t] = d0 * d0 + d1 * d1 + d2 * d2;
  __syncthreads();
  for (int o = 64; o > 0; o >>= 1) { if (t < o) red[t] += red[t + o]; __syncthreads(); }
  const float rs = rsqrtf(red[0] * (1.0f / DIM_) + 1e-6f);
  __syncthreads();

  const float v0 = d0 * rs * lnw[t] + lnb[t];
  const float v1 = d1 * rs * lnw[t + 128] + lnb[t + 128];
  const float v2 = d2 * rs * lnw[t + 256] + lnb[t + 256];
  red[t] = v0 * v0 + v1 * v1 + v2 * v2;
  __syncthreads();
  for (int o = 64; o > 0; o >>= 1) { if (t < o) red[t] += red[t + o]; __syncthreads(); }
  const float inv = 1.0f / fmaxf(sqrtf(red[0]), 1e-12f);

  _Float16* o = QK + row * DIM_;
  o[t] = (_Float16)(v0 * inv);
  o[t + 128] = (_Float16)(v1 * inv);
  o[t + 256] = (_Float16)(v2 * inv);
}

// ---------------- per-patch class histogram -> f16 label matrix -------------
__global__ void k_hist(const int* __restrict__ labels, _Float16* __restrict__ LAB) {
  __shared__ float hist[NC_];
  const int t = threadIdx.x;
  const int blk = blockIdx.x;            // b*256 + patch
  const int b = blk >> 8;
  const int patch = blk & 255;
  const int gy = patch >> 4, gx = patch & 15;
  for (int i = t; i < NC_; i += 256) hist[i] = 0.f;
  __syncthreads();
  if (t < PW_ * PW_) {
    const int py = t / PW_, px = t - py * PW_;
    const int y = gy * PW_ + py, x = gx * PW_ + px;
    const int cls = labels[(size_t)b * IMG_ * IMG_ + y * IMG_ + x];
    atomicAdd(&hist[cls], 1.0f);
  }
  __syncthreads();
  _Float16* o = LAB + (size_t)blk * NC_;
  for (int i = t; i < NC_; i += 256) o[i] = (_Float16)(hist[i] * (1.0f / 196.0f));
}

// ---- permuted + transposed + zero-padded labels: LABT[b][c:1024][k:2048] ----
// Built once so the attn@label GEMM streams contiguous B columns.
__global__ void k_labt(const _Float16* __restrict__ LAB, const int* __restrict__ perms,
                       _Float16* __restrict__ LABT) {
  const size_t tid = (size_t)blockIdx.x * 256 + threadIdx.x;
  const int k = (int)(tid & (NKEY_ - 1));
  const int c = (int)((tid >> 11) & (NCP_ - 1));
  const int b = (int)(tid >> 21);
  const int j = k >> 8, p = k & 255;
  const int bb = (j == 0) ? b : perms[(j - 1) * BS_ + b];
  _Float16 v = (_Float16)0.f;
  if (c < NC_) v = LAB[(size_t)(bb * NPATCH_ + p) * NC_ + c];
  LABT[tid] = v;
}

// ---------------- S = (q . k^T) / beta ----------------
// grid (8, 16, 4), block 128. Wave computes M=16, N=64.
__global__ void k_qk(const _Float16* __restrict__ QK, const int* __restrict__ perms,
                     float* __restrict__ S) {
  const int lane = threadIdx.x & 31;
  const int wave = threadIdx.x >> 5;
  const int b = blockIdx.z;
  const int n0 = (blockIdx.x * 4 + wave) * 64;
  const int m0 = blockIdx.y * 16;
  const _Float16* Arow = QK + (size_t)(b * NPATCH_ + m0 + (lane & 15)) * DIM_;
  const _Float16* key[4];
#pragma unroll
  for (int t = 0; t < 4; ++t) {
    const int nI = n0 + t * 16 + (lane & 15);
    const int j = nI >> 8, p = nI & 255;
    const int bb = (j == 0) ? b : perms[(j - 1) * BS_ + b];
    key[t] = QK + (size_t)(BS_ * NPATCH_ + bb * NPATCH_ + p) * DIM_;
  }
  v8f acc0 = {0.f,0.f,0.f,0.f,0.f,0.f,0.f,0.f};
  v8f acc1 = acc0, acc2 = acc0, acc3 = acc0;
#pragma unroll
  for (int k0 = 0; k0 < DIM_; k0 += 32) {
    v16h a = frag_a(Arow, k0, lane);
    WMMA(acc0, a, frag_b(key[0], k0, lane));
    WMMA(acc1, a, frag_b(key[1], k0, lane));
    WMMA(acc2, a, frag_b(key[2], k0, lane));
    WMMA(acc3, a, frag_b(key[3], k0, lane));
  }
  const int mB = m0 + ((lane >> 4) << 3);
  float* out = S + (size_t)b * NPATCH_ * NKEY_;
  const v8f* accs[4] = {&acc0, &acc1, &acc2, &acc3};
#pragma unroll
  for (int t = 0; t < 4; ++t) {
    const int nI = n0 + t * 16 + (lane & 15);
#pragma unroll
    for (int i = 0; i < 8; ++i)
      out[(size_t)(mB + i) * NKEY_ + nI] = (*accs[t])[i] * INV_BETA;
  }
}

// ---------------- row softmax over 2048 -> f16 attn ----------------
__global__ void k_softmax(const float* __restrict__ S, _Float16* __restrict__ A) {
  __shared__ float red[256];
  const int t = threadIdx.x;
  const size_t row = blockIdx.x;         // 0..1023
  const float* s = S + row * NKEY_;
  float v[8];
  float mx = -1e30f;
#pragma unroll
  for (int i = 0; i < 8; ++i) { v[i] = s[t + i * 256]; mx = fmaxf(mx, v[i]); }
  red[t] = mx;
  __syncthreads();
  for (int o = 128; o > 0; o >>= 1) {
    if (t < o) red[t] = fmaxf(red[t], red[t + o]);
    __syncthreads();
  }
  mx = red[0];
  __syncthreads();
  float sum = 0.f;
#pragma unroll
  for (int i = 0; i < 8; ++i) { v[i] = __expf(v[i] - mx); sum += v[i]; }
  red[t] = sum;
  __syncthreads();
  for (int o = 128; o > 0; o >>= 1) {
    if (t < o) red[t] += red[t + o];
    __syncthreads();
  }
  const float inv = 1.0f / red[0];
  _Float16* o = A + row * NKEY_;
#pragma unroll
  for (int i = 0; i < 8; ++i) o[t + i * 256] = (_Float16)(v[i] * inv);
}

// ---------------- label_hat = attn @ lab_d ----------------
// Attn (256x2048 per b), LABT (b,1024,2048). grid (16, 16, 4), block 32.
__global__ void k_al(const _Float16* __restrict__ Attn, const _Float16* __restrict__ LABT,
                     float* __restrict__ LH) {
  const int lane = threadIdx.x & 31;
  const int b = blockIdx.z;
  const int n0 = blockIdx.x * 64;        // class tile (padded to 1024)
  const int m0 = blockIdx.y * 16;
  const _Float16* Arow = Attn + (size_t)(b * NPATCH_ + m0 + (lane & 15)) * NKEY_;
  const _Float16* B0 = LABT + ((size_t)b * NCP_ + n0 + (lane & 15)) * NKEY_;
  const _Float16* B1 = B0 + (size_t)16 * NKEY_;
  const _Float16* B2 = B0 + (size_t)32 * NKEY_;
  const _Float16* B3 = B0 + (size_t)48 * NKEY_;
  v8f acc0 = {0.f,0.f,0.f,0.f,0.f,0.f,0.f,0.f};
  v8f acc1 = acc0, acc2 = acc0, acc3 = acc0;
#pragma unroll 2
  for (int k0 = 0; k0 < NKEY_; k0 += 32) {
    v16h a = frag_a(Arow, k0, lane);
    WMMA(acc0, a, frag_b(B0, k0, lane));
    WMMA(acc1, a, frag_b(B1, k0, lane));
    WMMA(acc2, a, frag_b(B2, k0, lane));
    WMMA(acc3, a, frag_b(B3, k0, lane));
  }
  const int mB = m0 + ((lane >> 4) << 3);
  float* o = LH + (size_t)(b * NPATCH_) * NC_;
  const v8f* accs[4] = {&acc0, &acc1, &acc2, &acc3};
#pragma unroll
  for (int t = 0; t < 4; ++t) {
    const int cI = n0 + t * 16 + (lane & 15);
    if (cI < NC_) {
#pragma unroll
      for (int i = 0; i < 8; ++i)
        o[(size_t)(mB + i) * NC_ + cI] = (*accs[t])[i];
    }
  }
}

// ---------------- nearest upsample 16x16 -> 224x224 (the 803 MB write) ------
__global__ void k_upsample(const float* __restrict__ LH, float* __restrict__ out) {
  const size_t total = (size_t)BS_ * NC_ * IMG_ * GRID_;
  size_t tid = (size_t)blockIdx.x * blockDim.x + threadIdx.x;
  if (tid >= total) return;
  const int gx = (int)(tid & 15);
  size_t t2 = tid >> 4;
  const int y = (int)(t2 % IMG_); t2 /= IMG_;
  const int c = (int)(t2 % NC_);
  const int b = (int)(t2 / NC_);
  const int patch = (y / PW_) * GRID_ + gx;
  const float v = LH[(size_t)(b * NPATCH_ + patch) * NC_ + c];
  float* o = out + (((size_t)(b * NC_ + c) * IMG_) + y) * IMG_ + (size_t)gx * PW_;
#pragma unroll
  for (int i = 0; i < PW_; ++i) o[i] = v;
}

// ---------------- launcher ----------------
extern "C" void kernel_launch(void* const* d_in, const int* in_sizes, int n_in,
                              void* d_out, int out_size, void* d_ws, size_t ws_size,
                              hipStream_t stream) {
  (void)in_sizes; (void)n_in; (void)out_size; (void)ws_size;
  const float* feat_t = (const float*)d_in[0];
  const float* feat_r = (const float*)d_in[1];
  const int*   labels = (const int*)d_in[2];
  const int*   perms  = (const int*)d_in[3];
  const float* fc1w   = (const float*)d_in[4];
  const float* fc1b   = (const float*)d_in[5];
  const float* fc2w   = (const float*)d_in[6];
  const float* fc2b   = (const float*)d_in[7];
  const float* lnw    = (const float*)d_in[8];
  const float* lnb    = (const float*)d_in[9];
  float* out = (float*)d_out;

  char* w = (char*)d_ws;
  size_t off = 0;
  auto alloc = [&](size_t bytes) -> void* {
    void* p = w + off;
    off = (off + bytes + 255) & ~(size_t)255;
    return p;
  };
  _Float16* Xh   = (_Float16*)alloc((size_t)NROW_ * DIM_ * 2);           // 1.5 MB
  _Float16* W1T  = (_Float16*)alloc((size_t)HID_ * DIM_ * 2);            // 2 MB
  _Float16* W2T  = (_Float16*)alloc((size_t)DIM_ * HID_ * 2);            // 2 MB
  _Float16* H    = (_Float16*)alloc((size_t)NROW_ * HID_ * 2);           // 11 MB
  float*    Y    = (float*)alloc((size_t)NROW_ * DIM_ * 4);              // 3 MB
  _Float16* QK   = (_Float16*)alloc((size_t)NROW_ * DIM_ * 2);           // 1.5 MB
  _Float16* LAB  = (_Float16*)alloc((size_t)BS_ * NPATCH_ * NC_ * 2);    // 2 MB
  _Float16* LABT = (_Float16*)alloc((size_t)BS_ * NCP_ * NKEY_ * 2);     // 16.8 MB
  float*    S    = (float*)alloc((size_t)BS_ * NPATCH_ * NKEY_ * 4);     // 8 MB
  _Float16* ATT  = (_Float16*)alloc((size_t)BS_ * NPATCH_ * NKEY_ * 2);  // 4 MB
  float*    LH   = (float*)alloc((size_t)BS_ * NPATCH_ * NC_ * 4);       // 4 MB

  const int nX = BS_ * NPATCH_ * DIM_;       // 393216 per half
  const int nW = DIM_ * HID_;                // 1032192
  k_cvt<<<(nX + 255) / 256, 256, 0, stream>>>(feat_t, Xh, nX);
  k_cvt<<<(nX + 255) / 256, 256, 0, stream>>>(feat_r, Xh + nX, nX);
  k_cvt_t<<<(nW + 255) / 256, 256, 0, stream>>>(fc1w, W1T, DIM_, HID_);
  k_cvt_t<<<(nW + 255) / 256, 256, 0, stream>>>(fc2w, W2T, HID_, DIM_);

  k_fc1_gelu<<<dim3(21, NROW_ / 16), 64, 0, stream>>>(Xh, W1T, fc1b, H);
  k_fc2<<<dim3(6, NROW_ / 16), 32, 0, stream>>>(H, W2T, Y);
  k_ln_l2<<<NROW_, 128, 0, stream>>>(Y, fc2b, lnw, lnb, QK);
  k_hist<<<BS_ * NPATCH_, 256, 0, stream>>>(labels, LAB);
  const size_t nLT = (size_t)BS_ * NCP_ * NKEY_;
  k_labt<<<(unsigned)(nLT / 256), 256, 0, stream>>>(LAB, perms, LABT);
  k_qk<<<dim3(8, NPATCH_ / 16, BS_), 128, 0, stream>>>(QK, perms, S);
  k_softmax<<<BS_ * NPATCH_, 256, 0, stream>>>(S, ATT);
  k_al<<<dim3(NCP_ / 64, NPATCH_ / 16, BS_), 32, 0, stream>>>(ATT, LABT, LH);
  const size_t tot = (size_t)BS_ * NC_ * IMG_ * GRID_;
  k_upsample<<<(unsigned)((tot + 255) / 256), 256, 0, stream>>>(LH, out);
}
